// MetasurfaceGNN_22265110463145
// MI455X (gfx1250) — compile-verified, matrix operand order checked by reading, not run on hardware
//
#include <hip/hip_runtime.h>

// ---------------------------------------------------------------------------
// Types for CDNA5 WMMA (wave32): V_WMMA_F32_16X16X32_BF16
// ---------------------------------------------------------------------------
typedef __attribute__((ext_vector_type(16))) __bf16        v16bf;
typedef __attribute__((ext_vector_type(8)))  float         v8f;
typedef __attribute__((ext_vector_type(8)))  unsigned int  v8u;

union FragBF {
  v8u   u;
  v16bf bf;
};

// Pack two f32 -> packed bf16x2 (round to nearest even).
__device__ __forceinline__ unsigned int pack2_bf16(float a, float b) {
  unsigned int ua = __float_as_uint(a);
  unsigned int ub = __float_as_uint(b);
  ua += 0x7FFFu + ((ua >> 16) & 1u);
  ub += 0x7FFFu + ((ub >> 16) & 1u);
  return (ua >> 16) | (ub & 0xFFFF0000u);
}

// K index of the low element of VGPR pair v for this half of the wave
// (16-bit A 16x32 / B 32x16 layout: V0..3 = K{0..7}+8*half, V4..7 = K{16..23}+8*half).
__device__ __forceinline__ int kbase_of(int v, int hsel) {
  return ((v < 4) ? (2 * v) : (16 + 2 * (v - 4))) + 8 * hsel;
}

// ---------------------------------------------------------------------------
// Kernel 1: edge MLP (WMMA bf16) + message matvec + atomic scatter-add
//   h   = relu(edge_attr * w1 + b1)                (E x 32)
//   w_e = h @ w2 + b2                              (E x 96 = E x 6 x 16)
//   msg[e,o] = sum_i x[src(e),i] * w_e[e,i,o]      (E x 16)
//   agg[dst(e),o] += msg[e,o]
// Each wave handles 16 edges; one 16x16x32 bf16 WMMA per 16-col tile (6 tiles).
// h is staged in LDS as pre-packed bf16x2 K-pairs so the A fragment is read
// with two ds_load_b128 per lane (pairs [4h..4h+3] -> u[0..3], [8+4h..] -> u[4..7]).
// ---------------------------------------------------------------------------
#define EK_WAVES 8   // 256 threads/block

__global__ __launch_bounds__(256)
void edge_msg_kernel(const float* __restrict__ x,
                     const int*   __restrict__ edge_index,   // (2,E) int32
                     const float* __restrict__ edge_attr,    // (E,1)
                     const float* __restrict__ w1,           // (1,32)
                     const float* __restrict__ b1,           // (32)
                     const float* __restrict__ w2,           // (32,96)
                     const float* __restrict__ b2,           // (96)
                     float*       __restrict__ agg,          // (N,16), pre-zeroed
                     int E_) {
  __shared__ unsigned int s_hp [EK_WAVES][16][16];  // 16 edges x 16 bf16x2 K-pairs
  __shared__ float        s_x  [EK_WAVES][16][6];
  __shared__ int          s_dst[EK_WAVES][16];

  const int tid  = threadIdx.x;
  const int wv   = tid >> 5;
  const int lane = tid & 31;
  const int hsel = lane >> 4;   // which half of the wave
  const int col  = lane & 15;

  // Persistent B fragments: w2 (32x96) as six 32x16 bf16 tiles in WMMA layout.
  FragBF bfrag[6];
#pragma unroll
  for (int i = 0; i < 6; ++i) {
#pragma unroll
    for (int v = 0; v < 8; ++v) {
      const int k = kbase_of(v, hsel);
      const int n = i * 16 + col;
      bfrag[i].u[v] = pack2_bf16(w2[k * 96 + n], w2[(k + 1) * 96 + n]);
    }
  }
  float b2v[6];
#pragma unroll
  for (int i = 0; i < 6; ++i) b2v[i] = b2[i * 16 + col];

  const float2* __restrict__ x2 = (const float2*)x;  // rows are 24B -> 8B aligned

  const int groups = E_ >> 7;   // 128 edges per block-iteration (E divisible)
  for (int g = blockIdx.x; g < groups; g += gridDim.x) {
    const int ebase = (g << 7) + wv * 16;

    // ---- per-edge prep: each half-wave lane owns edge m=col ----
    {
      const int m = col;
      const int e = ebase + m;
      const float a = edge_attr[e];

      // 16 of the 32 hidden units, packed at the producer into bf16x2 pairs.
      float hv[16];
#pragma unroll
      for (int jj = 0; jj < 16; ++jj) {
        const int j = jj + 16 * hsel;
        hv[jj] = fmaxf(fmaf(a, w1[j], b1[j]), 0.0f);
      }
      uint4 p0, p1;
      p0.x = pack2_bf16(hv[0],  hv[1]);  p0.y = pack2_bf16(hv[2],  hv[3]);
      p0.z = pack2_bf16(hv[4],  hv[5]);  p0.w = pack2_bf16(hv[6],  hv[7]);
      p1.x = pack2_bf16(hv[8],  hv[9]);  p1.y = pack2_bf16(hv[10], hv[11]);
      p1.z = pack2_bf16(hv[12], hv[13]); p1.w = pack2_bf16(hv[14], hv[15]);
      *(uint4*)&s_hp[wv][m][8 * hsel]     = p0;   // pairs [8h .. 8h+3]
      *(uint4*)&s_hp[wv][m][8 * hsel + 4] = p1;   // pairs [8h+4 .. 8h+7]

      // Low half loads src + x row (3x float2); high half loads dst.
      if (hsel == 0) {
        const int src = edge_index[e];
#pragma unroll
        for (int i = 0; i < 3; ++i)
          *(float2*)&s_x[wv][m][2 * i] = x2[src * 3 + i];
      } else {
        s_dst[wv][m] = edge_index[E_ + e];
      }
      // Prefetch next group's edge_attr (global_prefetch_b8).
      const int gn = g + gridDim.x;
      if (gn < groups) __builtin_prefetch(&edge_attr[(gn << 7) + wv * 16 + m], 0, 3);
    }
    __syncthreads();

    // ---- A fragment: two 16B LDS loads of pre-packed pairs ----
    FragBF afrag;
    {
      const uint4 a0 = *(const uint4*)&s_hp[wv][col][4 * hsel];       // u[0..3]
      const uint4 a1 = *(const uint4*)&s_hp[wv][col][8 + 4 * hsel];   // u[4..7]
      afrag.u[0] = a0.x; afrag.u[1] = a0.y; afrag.u[2] = a0.z; afrag.u[3] = a0.w;
      afrag.u[4] = a1.x; afrag.u[5] = a1.y; afrag.u[6] = a1.z; afrag.u[7] = a1.w;
    }

    // ---- six WMMAs: acc_i[row m, col o] = w_e[edge m, i, o] (before +b2) ----
    v8f zacc = {};
    v8f acc[6];
#pragma unroll
    for (int i = 0; i < 6; ++i) {
      acc[i] = __builtin_amdgcn_wmma_f32_16x16x32_bf16(
          false, afrag.bf, false, bfrag[i].bf, (short)0, zacc, false, false);
    }

    // ---- message matvec against live accumulators + atomic scatter ----
#pragma unroll
    for (int v = 0; v < 8; ++v) {
      const int m = v + 8 * hsel;                  // edge row in C/D layout
      float msg = 0.0f;
#pragma unroll
      for (int i = 0; i < 6; ++i)
        msg = fmaf(s_x[wv][m][i], acc[i][v] + b2v[i], msg);
      unsafeAtomicAdd(&agg[s_dst[wv][m] * 16 + col], msg);  // global_atomic_add_f32
    }
    __syncthreads();
  }
}

// ---------------------------------------------------------------------------
// Kernel 2: node update + crop into (B,16,5,5) image layout
//   out = relu(agg + x @ w_root + bias); keep 9x9 grid rows/cols 2..6.
// ---------------------------------------------------------------------------
__global__ __launch_bounds__(256)
void node_update_kernel(const float* __restrict__ x,
                        const float* __restrict__ w_root,  // (6,16)
                        const float* __restrict__ bias,    // (16)
                        const float* __restrict__ agg,     // (N,16)
                        float*       __restrict__ img,     // (B,16,5,5)
                        int N_) {
  const int t = blockIdx.x * blockDim.x + threadIdx.x;
  if (t >= N_ * 16) return;
  const int n = t >> 4;
  const int o = t & 15;
  float v = agg[t];
#pragma unroll
  for (int i = 0; i < 6; ++i) v = fmaf(x[n * 6 + i], w_root[i * 16 + o], v);
  v = fmaxf(v + bias[o], 0.0f);
  const int p = n % 81, b = n / 81;
  const int r = p / 9, c = p % 9;
  if (r >= 2 && r < 7 && c >= 2 && c < 7)
    img[(b * 16 + o) * 25 + (r - 2) * 5 + (c - 2)] = v;
}

// ---------------------------------------------------------------------------
// Kernel 3: fully-unrolled 6-layer 3x3 SAME conv stack on 5x5 tiles.
// One block per image; LDS ping-pong of 7x7 zero-padded activations;
// per-layer weights staged in LDS.
// ---------------------------------------------------------------------------
__global__ __launch_bounds__(256)
void conv_stack_kernel(const float* __restrict__ img,
                       const float* __restrict__ cw0, const float* __restrict__ cb0,
                       const float* __restrict__ cw1, const float* __restrict__ cb1,
                       const float* __restrict__ cw2, const float* __restrict__ cb2,
                       const float* __restrict__ cw3, const float* __restrict__ cb3,
                       const float* __restrict__ cw4, const float* __restrict__ cb4,
                       const float* __restrict__ cw5, const float* __restrict__ cb5,
                       float* __restrict__ out) {
  __shared__ float sbuf[2][16 * 49];   // 16 ch x 7x7 (zero-padded ring)
  __shared__ float swt[16 * 16 * 9];   // one layer's weights

  const float* cwp[6] = {cw0, cw1, cw2, cw3, cw4, cw5};
  const float* cbp[6] = {cb0, cb1, cb2, cb3, cb4, cb5};

  const int b = blockIdx.x;
  const int t = threadIdx.x;

  for (int i = t; i < 16 * 49; i += 256) { sbuf[0][i] = 0.0f; sbuf[1][i] = 0.0f; }
  __syncthreads();
  for (int i = t; i < 400; i += 256) {
    const int ch = i / 25, pos = i % 25, y = pos / 5, xx = pos % 5;
    sbuf[0][ch * 49 + (y + 1) * 7 + (xx + 1)] = img[b * 400 + i];
  }

#pragma unroll
  for (int l = 0; l < 6; ++l) {
    const int OC = (l < 5) ? 16 : 6;
    const float* __restrict__ cur = sbuf[l & 1];
    float* __restrict__       nxt = sbuf[(l + 1) & 1];
    __syncthreads();
    for (int i = t; i < OC * 16 * 9; i += 256) swt[i] = cwp[l][i];
    __syncthreads();
    for (int i = t; i < OC * 25; i += 256) {
      const int oc = i / 25, pos = i % 25, y = pos / 5, xx = pos % 5;
      float acc = cbp[l][oc];
#pragma unroll
      for (int ic = 0; ic < 16; ++ic) {
        const float* wp = &swt[(oc * 16 + ic) * 9];
        const float* ip = &cur[ic * 49 + y * 7 + xx];
#pragma unroll
        for (int dy = 0; dy < 3; ++dy)
#pragma unroll
          for (int dx = 0; dx < 3; ++dx)
            acc = fmaf(ip[dy * 7 + dx], wp[dy * 3 + dx], acc);
      }
      if (l < 5) nxt[oc * 49 + (y + 1) * 7 + (xx + 1)] = fmaxf(acc, 0.0f);
      else       out[b * 150 + i] = acc;
    }
  }
}

// ---------------------------------------------------------------------------
// Launch
// ---------------------------------------------------------------------------
extern "C" void kernel_launch(void* const* d_in, const int* in_sizes, int n_in,
                              void* d_out, int out_size, void* d_ws, size_t ws_size,
                              hipStream_t stream) {
  const float* x          = (const float*)d_in[0];
  const int*   edge_index = (const int*)  d_in[1];
  const float* edge_attr  = (const float*)d_in[2];
  const float* w1         = (const float*)d_in[3];
  const float* b1         = (const float*)d_in[4];
  const float* w2         = (const float*)d_in[5];
  const float* b2         = (const float*)d_in[6];
  const float* w_root     = (const float*)d_in[7];
  const float* bias       = (const float*)d_in[8];
  const float* cw0 = (const float*)d_in[9];  const float* cb0 = (const float*)d_in[10];
  const float* cw1 = (const float*)d_in[11]; const float* cb1 = (const float*)d_in[12];
  const float* cw2 = (const float*)d_in[13]; const float* cb2 = (const float*)d_in[14];
  const float* cw3 = (const float*)d_in[15]; const float* cb3 = (const float*)d_in[16];
  const float* cw4 = (const float*)d_in[17]; const float* cb4 = (const float*)d_in[18];
  const float* cw5 = (const float*)d_in[19]; const float* cb5 = (const float*)d_in[20];

  const int N_ = in_sizes[0] / 6;   // nodes
  const int E_ = in_sizes[2];       // edges (edge_attr is (E,1))
  const int B_ = N_ / 81;           // graphs

  float* agg = (float*)d_ws;                         // N x 16
  float* img = agg + (size_t)N_ * 16;                // B x 16 x 5 x 5

  (void)hipMemsetAsync(agg, 0, (size_t)N_ * 16 * sizeof(float), stream);
  // (note: workspace use = N*16*4 + B*400*4 bytes ~ 28 MB)

  const int groups  = E_ >> 7;
  const int eblocks = groups < 2048 ? groups : 2048;
  edge_msg_kernel<<<eblocks, 256, 0, stream>>>(x, edge_index, edge_attr,
                                               w1, b1, w2, b2, agg, E_);

  const int tot = N_ * 16;
  node_update_kernel<<<(tot + 255) / 256, 256, 0, stream>>>(x, w_root, bias,
                                                            agg, img, N_);

  conv_stack_kernel<<<B_, 256, 0, stream>>>(img,
                                            cw0, cb0, cw1, cb1, cw2, cb2,
                                            cw3, cb3, cw4, cb4, cw5, cb5,
                                            (float*)d_out);
}
